// ModernNCA_79310866088152
// MI455X (gfx1250) — matrix operand order repaired and checked by reading, not compiled, not run on hardware
//
#include <hip/hip_runtime.h>

// ---------------------------------------------------------------------------
// ModernNCA forward for MI455X (gfx1250, wave32, WMMA f32_16x16x32_f16)
// ---------------------------------------------------------------------------
typedef _Float16 half8  __attribute__((ext_vector_type(8)));
typedef _Float16 half16 __attribute__((ext_vector_type(16)));
typedef float    float8 __attribute__((ext_vector_type(8)));

#define NCAND  100000
#define NPAD   100016        // NCAND padded to a multiple of 16
#define NQ     1024
#define DNUM   64
#define DCAT   32
#define DIN    96
#define DIM    512
#define DOUTC  10
#define SEGW   2500          // half of the 5000-wide lse chunk
#define NSEG   40
#define NCHUNK 20

static __device__ __forceinline__ float8 wmma16(half16 a, half16 b, float8 c) {
  return __builtin_amdgcn_wmma_f32_16x16x32_f16(false, a, false, b, (short)0, c,
                                                false, false);
}
static __device__ __forceinline__ half16 join16(half8 lo, half8 hi) {
  return __builtin_shufflevector(lo, hi, 0, 1, 2, 3, 4, 5, 6, 7, 8, 9, 10, 11,
                                 12, 13, 14, 15);
}
static __device__ __forceinline__ half8 ldh8(const _Float16* p) {
  return *(const half8*)p;
}
static __device__ __forceinline__ half8 cvt8(const float* __restrict__ p) {
  half8 r;
#pragma unroll
  for (int i = 0; i < 8; ++i) r[i] = (_Float16)p[i];
  return r;
}

// ---------------------------------------------------------------------------
// Phase 0: pre-transpose W1 [96x512] and W2 [512x512] into f16 row-major
//          transposes (W1T[512][96], W2T[512][512]) so WMMA B-fragments are
//          two contiguous 8-half runs per lane (at koff and koff+16).
// ---------------------------------------------------------------------------
__global__ void prep_weights(const float* __restrict__ W1,
                             const float* __restrict__ W2,
                             _Float16* __restrict__ W1T,
                             _Float16* __restrict__ W2T) {
  int idx = blockIdx.x * blockDim.x + threadIdx.x;
  if (idx < DIM * DIN) {
    int n = idx / DIN, k = idx % DIN;
    W1T[idx] = (_Float16)W1[k * DIM + n];
  } else {
    int i2 = idx - DIM * DIN;
    if (i2 < DIM * DIM) {
      int n = i2 / DIM, k = i2 % DIM;
      W2T[i2] = (_Float16)W2[k * DIM + n];
    }
  }
}

// ---------------------------------------------------------------------------
// Phase 1: encode rows (candidates then queries) through the 2-layer MLP.
// One wave per 16-row tile. GEMM1 (K=96, 3 wmma k-steps) -> relu -> stage in
// LDS -> GEMM2 (K=512, 16 wmma k-steps) -> f16 embedding + squared norm.
// ---------------------------------------------------------------------------
__global__ __launch_bounds__(128) void encode_rows(
    const float* __restrict__ xnum, const float* __restrict__ xcat,
    const float* __restrict__ cnum, const float* __restrict__ ccat,
    const _Float16* __restrict__ W1T, const _Float16* __restrict__ W2T,
    const float* __restrict__ b1, const float* __restrict__ b2,
    _Float16* __restrict__ Ch, _Float16* __restrict__ Qh,
    float* __restrict__ cnorm, float* __restrict__ qnorm) {
  __shared__ __align__(16) _Float16 hbuf[4][16 * DIM];  // 64 KB: 16x512 f16 per wave

  const int wave = threadIdx.x >> 5;
  const int lane = threadIdx.x & 31;
  const int totalTiles = (NPAD + NQ) / 16;
  const int tile = blockIdx.x * 4 + wave;
  if (tile >= totalTiles) return;   // wave-uniform exit, EXEC stays full elsewhere

  const int  rowBase = tile * 16;
  const bool isCand  = rowBase < NPAD;
  const int  m       = lane & 15;          // A row / B,D column for this lane
  const bool hiHalf  = lane >= 16;
  const int  koff    = hiHalf ? 8 : 0;     // A-layout K interleave
  const int  kloB    = hiHalf ? 16 : 0;    // unused; B uses the same koff interleave as A
  const int  dOff    = hiHalf ? 8 : 0;     // D-layout row offset

  // ---- input A fragments: 16x96 tile, fp32 -> f16 on the fly --------------
  long srow;
  const float *np, *cp;
  if (isCand) {
    srow = rowBase + m;
    if (srow >= NCAND) srow = NCAND - 1;   // pad rows duplicate last candidate
    np = cnum; cp = ccat;
  } else {
    srow = rowBase + m - NPAD;
    np = xnum; cp = xcat;
  }
  half16 ain[3];
#pragma unroll
  for (int kt = 0; kt < 3; ++kt) {
    const int f0 = kt * 32 + koff;
    const int f1 = kt * 32 + 16 + koff;
    const float* p0 = (f0 < DNUM) ? (np + srow * DNUM + f0)
                                  : (cp + srow * DCAT + (f0 - DNUM));
    const float* p1 = (f1 < DNUM) ? (np + srow * DNUM + f1)
                                  : (cp + srow * DCAT + (f1 - DNUM));
    ain[kt] = join16(cvt8(p0), cvt8(p1));
  }

  _Float16* hrow = &hbuf[wave][0];

  // ---- GEMM1: h = relu(x @ W1 + b1) -> LDS (D layout scattered) -----------
#pragma unroll 1
  for (int nt = 0; nt < DIM / 16; ++nt) {
    const int col = nt * 16 + m;
    float8 acc = {};
#pragma unroll
    for (int kt = 0; kt < 3; ++kt) {
      const _Float16* bp = W1T + (size_t)col * DIN + kt * 32 + koff;
      acc = wmma16(ain[kt], join16(ldh8(bp), ldh8(bp + 16)), acc);
    }
    const float bias = b1[col];
#pragma unroll
    for (int r = 0; r < 8; ++r) {
      float v = acc[r] + bias;
      v = v > 0.f ? v : 0.f;
      hrow[(r + dOff) * DIM + col] = (_Float16)v;
    }
  }
  asm volatile("s_wait_dscnt 0" ::: "memory");   // cross-lane LDS RAW fence

  // ---- re-gather h as A fragments (16 k-steps of 32) ----------------------
  half16 ah[16];
#pragma unroll
  for (int kt = 0; kt < 16; ++kt) {
    const _Float16* p = hrow + m * DIM + kt * 32 + koff;
    ah[kt] = join16(ldh8(p), ldh8(p + 16));
  }

  // ---- GEMM2: c = h @ W2 + b2 -> f16 embedding + ||c||^2 ------------------
  float nacc[8] = {0, 0, 0, 0, 0, 0, 0, 0};
  const long dRow0 = rowBase + dOff;
  _Float16* dstBase = isCand ? (Ch + (size_t)dRow0 * DIM)
                             : (Qh + (size_t)(dRow0 - NPAD) * DIM);
#pragma unroll 1
  for (int nt = 0; nt < DIM / 16; ++nt) {
    const int col = nt * 16 + m;
    float8 acc = {};
#pragma unroll
    for (int kt = 0; kt < 16; ++kt) {
      const _Float16* bp = W2T + (size_t)col * DIM + kt * 32 + koff;
      acc = wmma16(ah[kt], join16(ldh8(bp), ldh8(bp + 16)), acc);
    }
    const float bias = b2[col];
#pragma unroll
    for (int r = 0; r < 8; ++r) {
      const float v = acc[r] + bias;
      dstBase[(size_t)r * DIM + col] = (_Float16)v;
      nacc[r] += v * v;
    }
  }
  // reduce row norms across the 16 column-lanes of each half-wave
#pragma unroll
  for (int r = 0; r < 8; ++r) {
    float v = nacc[r];
    v += __shfl_xor(v, 1, 32);
    v += __shfl_xor(v, 2, 32);
    v += __shfl_xor(v, 4, 32);
    v += __shfl_xor(v, 8, 32);
    nacc[r] = v;
  }
  if ((lane & 15) == 0) {
#pragma unroll
    for (int r = 0; r < 8; ++r) {
      const long row = rowBase + dOff + r;
      if (isCand) cnorm[row] = nacc[r];
      else        qnorm[row - NPAD] = nacc[r];
    }
  }
}

// ---------------------------------------------------------------------------
// Phase 2: S = Q @ C^T per 16x16 tile, then dist/exp epilogue with per-chunk
// sum-of-exp(ed) (for the chunked logsumexp) and class-bucketed ed sums.
// Grid: x = query tile (64), y = 2500-wide candidate segment (40).
// Consecutive blocks share a C segment -> L2-resident (192 MB >> 102 MB).
// ---------------------------------------------------------------------------
__global__ __launch_bounds__(256) void nca_distance(
    const _Float16* __restrict__ Qh, const _Float16* __restrict__ Ch,
    const float* __restrict__ qnorm, const float* __restrict__ cnorm,
    const int* __restrict__ cy, float* __restrict__ chunkSum,
    float* __restrict__ logitsG) {
  __shared__ __align__(16) _Float16 qbuf[16 * DIM];   // 16 KB query tile
  __shared__ float logitsS[16 * DOUTC];
  __shared__ float sumExpS[16];

  const int it = blockIdx.x;
  const int seg = blockIdx.y;
  const int ibase = it * 16;

  {  // cooperative copy of the 16x512 f16 query tile into LDS
    const uint4* src = (const uint4*)(Qh + (size_t)ibase * DIM);
    uint4* dst = (uint4*)qbuf;
    for (int idx = threadIdx.x; idx < (16 * DIM) / 8; idx += blockDim.x)
      dst[idx] = src[idx];
  }
  for (int idx = threadIdx.x; idx < 16 * DOUTC; idx += blockDim.x)
    logitsS[idx] = 0.f;
  if (threadIdx.x < 16) sumExpS[threadIdx.x] = 0.f;
  __syncthreads();

  const int  wave   = threadIdx.x >> 5;
  const int  lane   = threadIdx.x & 31;
  const int  m      = lane & 15;
  const bool hiHalf = lane >= 16;
  const int  koff   = hiHalf ? 8 : 0;
  const int  kloB   = hiHalf ? 16 : 0;
  const int  dOff   = hiHalf ? 8 : 0;

  half16 aq[16];
#pragma unroll
  for (int kt = 0; kt < 16; ++kt) {
    const _Float16* p = qbuf + m * DIM + kt * 32 + koff;
    aq[kt] = join16(ldh8(p), ldh8(p + 16));
  }
  float qn[8];
#pragma unroll
  for (int r = 0; r < 8; ++r) qn[r] = qnorm[ibase + dOff + r];

  float sacc[8] = {0, 0, 0, 0, 0, 0, 0, 0};
  const int jseg0 = seg * SEGW;
  const int jsegEnd = jseg0 + SEGW;
  const int ntiles = (SEGW + 15) / 16;   // 157 (last tile partially masked)

  for (int t = wave; t < ntiles; t += 8) {
    const int jbase = jseg0 + t * 16;
    const int j = jbase + m;                       // this lane's candidate row
    const int tn = t + 8;                          // prefetch the next C tile
    if (tn < ntiles)
      __builtin_prefetch(Ch + (size_t)(jseg0 + tn * 16 + m) * DIM, 0, 0);

    const _Float16* crow = Ch + (size_t)j * DIM;   // column n of B == row j of C
    float8 acc = {};
#pragma unroll
    for (int kt = 0; kt < 16; ++kt) {
      const _Float16* bp = crow + kt * 32 + koff;
      acc = wmma16(aq[kt], join16(ldh8(bp), ldh8(bp + 16)), acc);
    }
    if (j < jsegEnd) {                             // mask the segment overhang
      const float cn = cnorm[j];
      const int cls = cy[j];
#pragma unroll
      for (int r = 0; r < 8; ++r) {
        float sq = qn[r] + cn - 2.f * acc[r];
        sq = sq > 0.f ? sq : 0.f;
        const float e = __expf(-sqrtf(sq));        // ed = exp(-dist)
        sacc[r] += __expf(e);                      // for chunked logsumexp(ed)
        atomicAdd(&logitsS[(dOff + r) * DOUTC + cls], e);   // ed @ one_hot(y)
      }
    }
  }
#pragma unroll
  for (int r = 0; r < 8; ++r) {
    float v = sacc[r];
    v += __shfl_xor(v, 1, 32);
    v += __shfl_xor(v, 2, 32);
    v += __shfl_xor(v, 4, 32);
    v += __shfl_xor(v, 8, 32);
    if ((lane & 15) == 0) atomicAdd(&sumExpS[dOff + r], v);
  }
  __syncthreads();

  const int td = threadIdx.x;
  if (td < 16 * DOUTC) {
    atomicAdd(&logitsG[(size_t)(ibase + td / DOUTC) * DOUTC + (td % DOUTC)],
              logitsS[td]);
  } else if (td < 16 * DOUTC + 16) {
    const int r = td - 16 * DOUTC;
    atomicAdd(&chunkSum[(size_t)(ibase + r) * NCHUNK + (seg >> 1)], sumExpS[r]);
  }
}

// ---------------------------------------------------------------------------
// Phase 3: out[i][k] = log(logits[i][k]) - sum_c log(chunkSum[i][c])
// ---------------------------------------------------------------------------
__global__ void nca_finalize(const float* __restrict__ chunkSum,
                             const float* __restrict__ logitsG,
                             float* __restrict__ out) {
  const int idx = blockIdx.x * blockDim.x + threadIdx.x;
  if (idx >= NQ * DOUTC) return;
  const int i = idx / DOUTC;
  float lse = 0.f;
#pragma unroll
  for (int c = 0; c < NCHUNK; ++c) lse += __logf(chunkSum[i * NCHUNK + c]);
  out[idx] = __logf(logitsG[idx]) - lse;
}

// ---------------------------------------------------------------------------
extern "C" void kernel_launch(void* const* d_in, const int* in_sizes, int n_in,
                              void* d_out, int out_size, void* d_ws,
                              size_t ws_size, hipStream_t stream) {
  (void)in_sizes; (void)n_in; (void)out_size; (void)ws_size;
  const float* xnum = (const float*)d_in[0];
  const float* xcat = (const float*)d_in[1];
  const float* cnum = (const float*)d_in[2];
  const float* ccat = (const float*)d_in[3];
  const int*   cy   = (const int*)d_in[4];
  const float* W1   = (const float*)d_in[5];
  const float* b1   = (const float*)d_in[6];
  const float* W2   = (const float*)d_in[7];
  const float* b2   = (const float*)d_in[8];

  char* ws = (char*)d_ws;
  size_t off = 0;
  auto take = [&](size_t bytes) -> char* {
    char* r = ws + off;
    off = (off + bytes + 255) & ~(size_t)255;
    return r;
  };
  _Float16* W1T   = (_Float16*)take((size_t)DIM * DIN * 2);
  _Float16* W2T   = (_Float16*)take((size_t)DIM * DIM * 2);
  _Float16* Ch    = (_Float16*)take((size_t)NPAD * DIM * 2);
  _Float16* Qh    = (_Float16*)take((size_t)NQ * DIM * 2);
  float* cnorm    = (float*)take((size_t)NPAD * 4);
  float* qnorm    = (float*)take((size_t)NQ * 4);
  float* chunkSum = (float*)take((size_t)NQ * NCHUNK * 4);  // 81920 B (256-mult)
  float* logitsG  = (float*)take((size_t)NQ * DOUTC * 4);   // contiguous after

  prep_weights<<<(DIM * DIN + DIM * DIM + 255) / 256, 256, 0, stream>>>(
      W1, W2, W1T, W2T);
  hipMemsetAsync(chunkSum, 0, (size_t)NQ * (NCHUNK + DOUTC) * 4, stream);

  const int totalTiles = (NPAD + NQ) / 16;   // 6315 sixteen-row tiles
  encode_rows<<<(totalTiles + 3) / 4, 128, 0, stream>>>(
      xnum, xcat, cnum, ccat, W1T, W2T, b1, b2, Ch, Qh, cnorm, qnorm);

  dim3 g2(NQ / 16, NSEG);
  nca_distance<<<g2, 256, 0, stream>>>(Qh, Ch, qnorm, cnorm, cy, chunkSum,
                                       logitsG);
  nca_finalize<<<(NQ * DOUTC + 255) / 256, 256, 0, stream>>>(chunkSum, logitsG,
                                                             (float*)d_out);
}
